// CachedMultiHeadAttention_52407190946373
// MI455X (gfx1250) — matrix operand order repaired and checked by reading, not compiled, not run on hardware
//
#include <hip/hip_runtime.h>

// ---------------------------------------------------------------------------
// CDNA5 (gfx1250) causal multi-head attention:
//   Q/K/V projections (bf16 WMMA GEMM) -> flash attention (WMMA QK^T + PV,
//   async global->LDS staging, double-buffered) -> output projection.
// B=4, T=2048, D=1024, H=16, HD=64.
// Matrix math: v_wmma_f32_16x16x32_bf16.  Copies: global_load_async_to_lds_b128.
// ---------------------------------------------------------------------------

typedef __attribute__((ext_vector_type(16))) __bf16        v16bf;
typedef __attribute__((ext_vector_type(8)))  float         v8f;
typedef __attribute__((ext_vector_type(4)))  unsigned int  u32x4;
typedef __attribute__((ext_vector_type(2)))  unsigned int  u32x2;
typedef __attribute__((ext_vector_type(4)))  float         f32x4;

union FragU {
    u32x4          q[2];
    v16bf          v;
    unsigned short s[16];
};

__device__ __forceinline__ unsigned short f2bf(float f) {
    unsigned int u = __builtin_bit_cast(unsigned int, f);
    return (unsigned short)((u + 0x7FFFu + ((u >> 16) & 1u)) >> 16);
}

__device__ __forceinline__ v8f wmma_bf16(v16bf a, v16bf b, v8f c) {
    return __builtin_amdgcn_wmma_f32_16x16x32_bf16(false, a, false, b,
                                                   (short)0, c, false, false);
}

// Async DMA of 16 bytes, global -> LDS, per lane (ISA 15.18.3, ASYNCcnt).
// LDS address operand = low 32 bits of the generic pointer (wave-relative
// LDS byte offset after the AS3->generic addrspacecast).
__device__ __forceinline__ void async_copy_b128(void* lds_dst, const void* gsrc) {
    const unsigned      loff = (unsigned)(size_t)lds_dst;
    const unsigned long long ga = (unsigned long long)(size_t)gsrc;
    asm volatile("global_load_async_to_lds_b128 %0, %1, off"
                 :: "v"(loff), "v"(ga)
                 : "memory");
}

__device__ __forceinline__ void wait_async0() {
    asm volatile("s_wait_asynccnt 0x0" ::: "memory");
}

// A fragment (16x32 bf16, M x K): lane<16 -> row, halves j<8: K=(lane>=16)*8+j,
// j>=8: +16.  Two 16B LDS chunks per lane.
__device__ __forceinline__ v16bf frag_a_ld(const unsigned short* base, int rowBase,
                                           int stride, int koff, int lane) {
    const int row = rowBase + (lane & 15);
    const int k0  = koff + ((lane >> 4) << 3);
    const unsigned short* p = base + row * stride + k0;
    FragU f;
    f.q[0] = *reinterpret_cast<const u32x4*>(p);
    f.q[1] = *reinterpret_cast<const u32x4*>(p + 16);
    return f.v;
}

// B fragment (32x16 bf16, K x N): lane col = lane&15, halves j<8: K=(lane>=16)*8+j,
// j>=8: +16.
// Two 16B LDS chunks per lane (LDS stores B row-major as [n][k]).
__device__ __forceinline__ v16bf frag_b_ld(const unsigned short* base, int nBase,
                                           int stride, int koff, int lane) {
    const int n  = nBase + (lane & 15);
    const int k0 = koff + ((lane >> 4) << 3);
    const unsigned short* p = base + n * stride + k0;
    FragU f;
    f.q[0] = *reinterpret_cast<const u32x4*>(p);
    f.q[1] = *reinterpret_cast<const u32x4*>(p + 16);
    return f.v;
}

// ---------------------------------------------------------------------------
// GEMM with bias: Y[M,N] = A[M,K] @ W[N,K]^T + bias[N]
// 256 threads (8 waves), tile 128(M) x 64(N), K step 32; wave grid 4x2,
// each wave 32x32 via 2x2 WMMA tiles.  bf16 A tiles stream via async DMA.
// ---------------------------------------------------------------------------
template <bool A_BF16, bool OUT_BF16>
__global__ __launch_bounds__(256) void wmma_gemm_bias_kernel(
    const void* __restrict__ Ap, const float* __restrict__ W,
    const float* __restrict__ bias, void* __restrict__ Yp,
    int M, int N, int K) {
    __shared__ __align__(16) unsigned short lA[128 * 32];
    __shared__ __align__(16) unsigned short lW[64 * 32];

    const int tid   = threadIdx.x;
    const int lane  = tid & 31;
    const int wave  = tid >> 5;
    const int wm    = wave >> 1;  // 0..3
    const int wn    = wave & 1;   // 0..1
    const int mBase = blockIdx.y * 128;
    const int nBase = blockIdx.x * 64;

    v8f acc[2][2] = {};

    const int kSteps = K >> 5;
    for (int ks = 0; ks < kSteps; ++ks) {
        const int kOff = ks << 5;

        // --- stage A tile (128 x 32) into LDS as bf16 ---
        if constexpr (A_BF16) {
            const unsigned short* A = (const unsigned short*)Ap;
#pragma unroll
            for (int i = 0; i < 2; ++i) {
                const int c   = tid + (i << 8);   // 512 chunks of 8 halves
                const int row = c >> 2;
                const int col = (c & 3) << 3;
                async_copy_b128(lA + row * 32 + col,
                                A + (size_t)(mBase + row) * K + kOff + col);
            }
        } else {
            const float* A = (const float*)Ap;
#pragma unroll
            for (int i = 0; i < 4; ++i) {
                const int c   = tid + (i << 8);   // 1024 chunks of 4 floats
                const int row = c >> 3;
                const int col = (c & 7) << 2;
                f32x4 d = *reinterpret_cast<const f32x4*>(
                    A + (size_t)(mBase + row) * K + kOff + col);
                u32x2 pk;
                pk.x = (unsigned)f2bf(d.x) | ((unsigned)f2bf(d.y) << 16);
                pk.y = (unsigned)f2bf(d.z) | ((unsigned)f2bf(d.w) << 16);
                *reinterpret_cast<u32x2*>(lA + row * 32 + col) = pk;
            }
        }

        // --- stage W tile (64 x 32) into LDS as bf16 (fp32 convert in VALU) ---
#pragma unroll
        for (int i = 0; i < 2; ++i) {
            const int c   = tid + (i << 8);       // 512 chunks of 4 floats
            const int row = c >> 3;
            const int col = (c & 7) << 2;
            f32x4 d = *reinterpret_cast<const f32x4*>(
                W + (size_t)(nBase + row) * K + kOff + col);
            u32x2 pk;
            pk.x = (unsigned)f2bf(d.x) | ((unsigned)f2bf(d.y) << 16);
            pk.y = (unsigned)f2bf(d.z) | ((unsigned)f2bf(d.w) << 16);
            *reinterpret_cast<u32x2*>(lW + row * 32 + col) = pk;
        }
        if constexpr (A_BF16) wait_async0();
        __syncthreads();

        v16bf af[2], bf_[2];
#pragma unroll
        for (int mt = 0; mt < 2; ++mt)
            af[mt] = frag_a_ld(lA, wm * 32 + mt * 16, 32, 0, lane);
#pragma unroll
        for (int nt = 0; nt < 2; ++nt)
            bf_[nt] = frag_b_ld(lW, wn * 32 + nt * 16, 32, 0, lane);
#pragma unroll
        for (int mt = 0; mt < 2; ++mt)
#pragma unroll
            for (int nt = 0; nt < 2; ++nt)
                acc[mt][nt] = wmma_bf16(af[mt], bf_[nt], acc[mt][nt]);
        __syncthreads();
    }

    // --- epilogue: bias add + store (D layout: VGPR i -> row i + 8*(lane>=16)) ---
#pragma unroll
    for (int mt = 0; mt < 2; ++mt) {
#pragma unroll
        for (int nt = 0; nt < 2; ++nt) {
            const int col = nBase + wn * 32 + nt * 16 + (lane & 15);
            const float bv = bias[col];
            const int rb = mBase + wm * 32 + mt * 16 + ((lane >> 4) << 3);
#pragma unroll
            for (int i = 0; i < 8; ++i) {
                const float y = acc[mt][nt][i] + bv;
                const size_t idx = (size_t)(rb + i) * N + col;
                if constexpr (OUT_BF16)
                    ((unsigned short*)Yp)[idx] = f2bf(y);
                else
                    ((float*)Yp)[idx] = y;
            }
        }
    }
}

// ---------------------------------------------------------------------------
// Flash attention, causal, per (b,h).  128 threads (4 waves), 64 query rows
// per workgroup (16 per wave), key blocks of 32 (= WMMA K for P@V).
// K tile and V staging are double-buffered and streamed with async DMA;
// prefetch of block sb+1 is issued before computing block sb.
// ---------------------------------------------------------------------------
__global__ __launch_bounds__(128) void flash_attn_kernel(
    const unsigned short* __restrict__ Q, const unsigned short* __restrict__ Kb,
    const unsigned short* __restrict__ V, unsigned short* __restrict__ O) {
    constexpr int T = 2048, D = 1024, HD = 64;

    __shared__ __align__(16) unsigned short lQ[64 * 64];        // 8 KB
    __shared__ __align__(16) unsigned short lK[2][32 * 64];     // 8 KB
    __shared__ __align__(16) unsigned short lVs[2][32 * 64];    // 8 KB staging
    __shared__ __align__(16) unsigned short lVT[64 * 32];       // 4 KB transposed
    __shared__ __align__(16) unsigned short lP[4][16 * 32];     // 4 KB

    const int tid  = threadIdx.x;
    const int lane = tid & 31;
    const int wave = tid >> 5;
    const int qb   = blockIdx.x * 64;
    const int b    = blockIdx.y >> 4;
    const int h    = blockIdx.y & 15;

    const size_t headOff = (size_t)b * T * D + (size_t)h * HD;

    // --- prologue: async-load Q tile (64x64) and block 0 of K and V ---
#pragma unroll
    for (int i = 0; i < 4; ++i) {
        const int c   = tid + (i << 7);  // 512 chunks of 8 halves
        const int row = c >> 3;
        const int col = (c & 7) << 3;
        async_copy_b128(lQ + row * 64 + col,
                        Q + headOff + (size_t)(qb + row) * D + col);
    }
#pragma unroll
    for (int i = 0; i < 2; ++i) {
        const int c   = tid + (i << 7);  // 256 chunks
        const int row = c >> 3;
        const int col = (c & 7) << 3;
        async_copy_b128(lK[0] + row * 64 + col,
                        Kb + headOff + (size_t)row * D + col);
        async_copy_b128(lVs[0] + row * 64 + col,
                        V + headOff + (size_t)row * D + col);
    }
    wait_async0();
    __syncthreads();

    // Q fragments for both 32-wide K(=hd) windows, kept in registers.
    v16bf aq[2];
#pragma unroll
    for (int kw = 0; kw < 2; ++kw)
        aq[kw] = frag_a_ld(lQ, wave * 16, 64, kw * 32, lane);

    float m_i[8], l_i[8];
    v8f acc_o[4] = {};
#pragma unroll
    for (int i = 0; i < 8; ++i) { m_i[i] = -3.0e38f; l_i[i] = 0.0f; }

    const int qRow0   = qb + wave * 16 + ((lane >> 4) << 3);
    const int nBlocks = (qb >> 5) + 2;  // causal: keys up to qb+63

    for (int sb = 0; sb < nBlocks; ++sb) {
        const int cur = sb & 1;
        const int nxt = cur ^ 1;

        // --- issue async prefetch of block sb+1 (overlaps with this block) ---
        if (sb + 1 < nBlocks) {
            const int sNext = (sb + 1) << 5;
#pragma unroll
            for (int i = 0; i < 2; ++i) {
                const int c   = tid + (i << 7);
                const int row = c >> 3;
                const int col = (c & 7) << 3;
                async_copy_b128(lK[nxt] + row * 64 + col,
                                Kb + headOff + (size_t)(sNext + row) * D + col);
                async_copy_b128(lVs[nxt] + row * 64 + col,
                                V + headOff + (size_t)(sNext + row) * D + col);
            }
        }

        // --- transpose staged V tile: lVs[cur][s][hd] -> lVT[hd][s] ---
#pragma unroll
        for (int i = 0; i < 2; ++i) {
            const int c   = tid + (i << 7);
            const int row = c >> 3;          // s
            const int col = (c & 7) << 3;    // hd
            FragU u;
            u.q[0] = *reinterpret_cast<const u32x4*>(lVs[cur] + row * 64 + col);
#pragma unroll
            for (int j = 0; j < 8; ++j)
                lVT[(col + j) * 32 + row] = u.s[j];
        }
        __syncthreads();

        const int sBase = sb << 5;

        // --- scores S = Q K^T / sqrt(HD), causal mask ---
        float sc[2][8];
#pragma unroll
        for (int nt = 0; nt < 2; ++nt) {
            v8f s = {};
#pragma unroll
            for (int kw = 0; kw < 2; ++kw) {
                v16bf bk = frag_b_ld(lK[cur], nt * 16, 64, kw * 32, lane);
                s = wmma_bf16(aq[kw], bk, s);
            }
            const int sPos = sBase + nt * 16 + (lane & 15);
#pragma unroll
            for (int i = 0; i < 8; ++i) {
                const float v = s[i] * 0.125f;  // 1/sqrt(64)
                sc[nt][i] = (sPos > qRow0 + i) ? -3.0e38f : v;
            }
        }

        // --- online softmax (each row lives in one 16-lane half) ---
        float p0[8], p1[8];
#pragma unroll
        for (int i = 0; i < 8; ++i) {
            float rmax = fmaxf(sc[0][i], sc[1][i]);
            rmax = fmaxf(rmax, __shfl_xor(rmax, 1));
            rmax = fmaxf(rmax, __shfl_xor(rmax, 2));
            rmax = fmaxf(rmax, __shfl_xor(rmax, 4));
            rmax = fmaxf(rmax, __shfl_xor(rmax, 8));
            const float mNew = fmaxf(m_i[i], rmax);
            const float corr = __expf(m_i[i] - mNew);
            m_i[i] = mNew;
            p0[i] = __expf(sc[0][i] - mNew);
            p1[i] = __expf(sc[1][i] - mNew);
            float rs = p0[i] + p1[i];
            rs += __shfl_xor(rs, 1);
            rs += __shfl_xor(rs, 2);
            rs += __shfl_xor(rs, 4);
            rs += __shfl_xor(rs, 8);
            l_i[i] = l_i[i] * corr + rs;
#pragma unroll
            for (int t = 0; t < 4; ++t) acc_o[t][i] *= corr;
        }

        // --- stage P (bf16) in this wave's LDS buffer ---
        unsigned short* pw = lP[wave];
        {
            const int r0 = (lane >> 4) << 3;
            const int cL = lane & 15;
#pragma unroll
            for (int i = 0; i < 8; ++i) {
                pw[(r0 + i) * 32 + cL]      = f2bf(p0[i]);
                pw[(r0 + i) * 32 + cL + 16] = f2bf(p1[i]);
            }
        }

        // --- O += P @ V (one WMMA per 16-wide hd tile, K = 32 keys) ---
        const v16bf ap = frag_a_ld(pw, 0, 32, 0, lane);
#pragma unroll
        for (int t = 0; t < 4; ++t) {
            v16bf bv = frag_b_ld(lVT, t * 16, 32, 0, lane);
            acc_o[t] = wmma_bf16(ap, bv, acc_o[t]);
        }

        // --- retire this iteration's prefetch before the buffers are used ---
        wait_async0();
        __syncthreads();
    }

    // --- normalize and write attention output (bf16 workspace) ---
    float inv[8];
#pragma unroll
    for (int i = 0; i < 8; ++i) inv[i] = 1.0f / l_i[i];
    const int rb = qb + wave * 16 + ((lane >> 4) << 3);
    const int cL = lane & 15;
#pragma unroll
    for (int t = 0; t < 4; ++t) {
#pragma unroll
        for (int i = 0; i < 8; ++i) {
            const float v = acc_o[t][i] * inv[i];
            O[headOff + (size_t)(rb + i) * D + t * 16 + cL] = f2bf(v);
        }
    }
}

// ---------------------------------------------------------------------------
extern "C" void kernel_launch(void* const* d_in, const int* in_sizes, int n_in,
                              void* d_out, int out_size, void* d_ws, size_t ws_size,
                              hipStream_t stream) {
    (void)in_sizes; (void)n_in; (void)out_size; (void)ws_size;
    constexpr int B = 4, T = 2048, D = 1024, H = 16;
    const int M = B * T;  // 8192

    const float* query = (const float*)d_in[0];
    const float* key_  = (const float*)d_in[1];
    const float* value = (const float*)d_in[2];
    // d_in[3] is the causal mask; computed analytically in-kernel.
    const float* Wq = (const float*)d_in[4];
    const float* bq = (const float*)d_in[5];
    const float* Wk = (const float*)d_in[6];
    const float* bk = (const float*)d_in[7];
    const float* Wv = (const float*)d_in[8];
    const float* bv = (const float*)d_in[9];
    const float* Wo = (const float*)d_in[10];
    const float* bo = (const float*)d_in[11];
    float* out = (float*)d_out;

    // Workspace: 4 bf16 [B*T, D] buffers = 64 MB total.
    unsigned short* Qbf = (unsigned short*)d_ws;
    unsigned short* Kbf = Qbf + (size_t)M * D;
    unsigned short* Vbf = Kbf + (size_t)M * D;
    unsigned short* Abf = Vbf + (size_t)M * D;

    dim3 gGemm(D / 64, M / 128);  // (16, 64)
    wmma_gemm_bias_kernel<false, true><<<gGemm, 256, 0, stream>>>(query, Wq, bq, Qbf, M, D, D);
    wmma_gemm_bias_kernel<false, true><<<gGemm, 256, 0, stream>>>(key_,  Wk, bk, Kbf, M, D, D);
    wmma_gemm_bias_kernel<false, true><<<gGemm, 256, 0, stream>>>(value, Wv, bv, Vbf, M, D, D);

    dim3 gAttn(T / 64, B * H);    // (32, 64)
    flash_attn_kernel<<<gAttn, 128, 0, stream>>>(Qbf, Kbf, Vbf, Abf);

    wmma_gemm_bias_kernel<true, false><<<gGemm, 256, 0, stream>>>(Abf, Wo, bo, out, M, D, D);
}
